// MultiScaleRetention_70531952935483
// MI455X (gfx1250) — compile-verified, matrix operand order checked by reading, not run on hardware
//
#include <hip/hip_runtime.h>
#include <math.h>

typedef __bf16 bf16_t;
typedef __attribute__((ext_vector_type(16))) __bf16 v16bf;
typedef __attribute__((ext_vector_type(8)))  __bf16 bf16x8;
typedef __attribute__((ext_vector_type(8)))  float  v8f;

#define B_  2
#define S_  4096
#define D_  1024
#define H_  16
#define DK_ 64
#define DV_ 128
#define NC_ 64   /* chunks */
#define CL_ 64   /* chunk length */
#define BS_ 8192
#define NPROJ_ 6144

__device__ __forceinline__ v8f wmma_bf16(v16bf a, v16bf b, v8f c) {
    return __builtin_amdgcn_wmma_f32_16x16x32_bf16(false, a, false, b, (short)0, c, false, false);
}

// A/B fragment, row-contiguous source: lane holds row (lane%16), kb=(lane/16)*8.
__device__ __forceinline__ v16bf load_frag_contig(const bf16_t* rowk0, int kb) {
    bf16x8 lo = *(const bf16x8*)(rowk0 + kb);
    bf16x8 hi = *(const bf16x8*)(rowk0 + kb + 16);
    return __builtin_shufflevector(lo, hi, 0,1,2,3,4,5,6,7,8,9,10,11,12,13,14,15);
}

__device__ __forceinline__ v16bf load_frag_contig_scaled(const bf16_t* rowk0, int kb, float s) {
    v16bf f; const bf16_t* p = rowk0 + kb;
#pragma unroll
    for (int i = 0; i < 8; ++i) {
        f[i]     = (bf16_t)((float)p[i] * s);
        f[i + 8] = (bf16_t)((float)p[i + 16] * s);
    }
    return f;
}

__device__ __forceinline__ v16bf load_frag_strided(const bf16_t* p0, int stride) {
    v16bf f;
#pragma unroll
    for (int i = 0; i < 8; ++i) {
        f[i]     = p0[(size_t)i * stride];
        f[i + 8] = p0[(size_t)(i + 16) * stride];
    }
    return f;
}

__device__ __forceinline__ v16bf load_frag_f32_scaled(const float* p0, float s) {
    v16bf f;
#pragma unroll
    for (int i = 0; i < 8; ++i) {
        f[i]     = (bf16_t)(p0[i] * s);
        f[i + 8] = (bf16_t)(p0[i + 16] * s);
    }
    return f;
}

// ---------------- K0: per-head retention constants ----------------
__global__ void prep_const_kernel(const float* __restrict__ am, float* __restrict__ mask_n,
                                  float* __restrict__ inner_decay, float* __restrict__ cdec) {
    const int h = blockIdx.x, i = threadIdx.x;       // 16 blocks x 64 threads
    float dec = logf(1.f - exp2f(-5.f - (float)h));
    float m[64]; float rs = 0.f;
#pragma unroll 8
    for (int j = 0; j < 64; ++j) {
        float e = (j <= i) ? expf(am[i * 64 + j] * dec) : 0.f;
        m[j] = e; rs += e;
    }
    float sc = sqrtf(rs);
    __shared__ float scs[64];
    scs[i] = sc;
    __syncthreads();
    float inv = 1.f / sc;
#pragma unroll 8
    for (int j = 0; j < 64; ++j) mask_n[(size_t)h * 4096 + i * 64 + j] = m[j] * inv;
    inner_decay[h * 64 + i] = expf(dec * (float)(i + 1)) * scs[63] * inv;
    if (i == 0) cdec[h] = expf(dec * (float)NC_);
}

// ---------------- pack helpers ----------------
__global__ void f32_to_bf16_kernel(const float* __restrict__ src, bf16_t* __restrict__ dst, size_t n) {
    size_t i = (size_t)blockIdx.x * blockDim.x + threadIdx.x;
    size_t stride = (size_t)gridDim.x * blockDim.x;
    for (; i < n; i += stride) dst[i] = (bf16_t)src[i];
}

__global__ void copy_f32_kernel(const float* __restrict__ src, float* __restrict__ dst, int n) {
    int i = blockIdx.x * blockDim.x + threadIdx.x;
    if (i < n) dst[i] = src[i];
}

// ---------------- K2: LayerNorm -> bf16 ----------------
__global__ void layernorm_kernel(const float* __restrict__ x, const float* __restrict__ g,
                                 const float* __restrict__ bta, bf16_t* __restrict__ xn) {
    const int row = blockIdx.x, t = threadIdx.x;     // 8192 blocks x 256
    const float* pr = x + (size_t)row * D_;
    float v[4], s = 0.f, q = 0.f;
#pragma unroll
    for (int i = 0; i < 4; ++i) { v[i] = pr[t * 4 + i]; s += v[i]; q += v[i] * v[i]; }
    __shared__ float ps[256], pq[256];
    ps[t] = s; pq[t] = q; __syncthreads();
    for (int o = 128; o > 0; o >>= 1) {
        if (t < o) { ps[t] += ps[t + o]; pq[t] += pq[t + o]; }
        __syncthreads();
    }
    float mu = ps[0] * (1.f / D_);
    float var = pq[0] * (1.f / D_) - mu * mu;
    float rsd = rsqrtf(var + 1e-5f);
    bf16_t* xr = xn + (size_t)row * D_;
#pragma unroll
    for (int i = 0; i < 4; ++i) {
        int ch = t * 4 + i;
        xr[ch] = (bf16_t)((v[i] - mu) * rsd * g[ch] + bta[ch]);
    }
}

// ---------------- K3: WMMA GEMM  C[M,N] = A[M,K] @ W[N,K]^T + bias ----------------
// Each wave computes a 32x64 tile: 2 A fragments reused over 4 B fragments
// (8 WMMAs per 6 fragment loads, ~21 FLOP/byte from cache).
__global__ void gemm_bf16_kernel(const bf16_t* __restrict__ A, const bf16_t* __restrict__ W,
                                 const float* __restrict__ bias, float* __restrict__ Cmat,
                                 int M, int N, int K) {
    const int lane = threadIdx.x & 31;
    const int wave = threadIdx.x >> 5;               // 8 waves
    const int n0 = blockIdx.x * 64;
    const int m0 = blockIdx.y * 256 + wave * 32;
    const int r = lane & 15;
    const int kb = (lane >> 4) << 3;
    const bf16_t* a0 = A + (size_t)(m0 + r) * K;
    const bf16_t* a1 = A + (size_t)(m0 + 16 + r) * K;
    const bf16_t* w0 = W + (size_t)(n0 + r) * K;
    const bf16_t* w1 = W + (size_t)(n0 + 16 + r) * K;
    const bf16_t* w2 = W + (size_t)(n0 + 32 + r) * K;
    const bf16_t* w3 = W + (size_t)(n0 + 48 + r) * K;
    v8f acc00 = {}, acc01 = {}, acc02 = {}, acc03 = {};
    v8f acc10 = {}, acc11 = {}, acc12 = {}, acc13 = {};
    for (int k = 0; k < K; k += 32) {
        // stream-ahead prefetch of activation rows (speculative; OOB silently dropped)
        __builtin_prefetch(a0 + k + 512, 0, 1);
        __builtin_prefetch(a1 + k + 512, 0, 1);
        v16bf aa0 = load_frag_contig(a0 + k, kb);
        v16bf aa1 = load_frag_contig(a1 + k, kb);
        v16bf b0  = load_frag_contig(w0 + k, kb);
        v16bf b1  = load_frag_contig(w1 + k, kb);
        v16bf b2  = load_frag_contig(w2 + k, kb);
        v16bf b3  = load_frag_contig(w3 + k, kb);
        acc00 = wmma_bf16(aa0, b0, acc00);
        acc10 = wmma_bf16(aa1, b0, acc10);
        acc01 = wmma_bf16(aa0, b1, acc01);
        acc11 = wmma_bf16(aa1, b1, acc11);
        acc02 = wmma_bf16(aa0, b2, acc02);
        acc12 = wmma_bf16(aa1, b2, acc12);
        acc03 = wmma_bf16(aa0, b3, acc03);
        acc13 = wmma_bf16(aa1, b3, acc13);
    }
    const int dn = lane & 15;
    const int mh = (lane >> 4) << 3;
#pragma unroll
    for (int g = 0; g < 8; ++g) {
        size_t row0 = (size_t)(m0 + mh + g) * N + n0 + dn;
        size_t row1 = (size_t)(m0 + 16 + mh + g) * N + n0 + dn;
        float bb0 = bias[n0 + dn],      bb1 = bias[n0 + 16 + dn];
        float bb2 = bias[n0 + 32 + dn], bb3 = bias[n0 + 48 + dn];
        Cmat[row0]      = acc00[g] + bb0;
        Cmat[row0 + 16] = acc01[g] + bb1;
        Cmat[row0 + 32] = acc02[g] + bb2;
        Cmat[row0 + 48] = acc03[g] + bb3;
        Cmat[row1]      = acc10[g] + bb0;
        Cmat[row1 + 16] = acc11[g] + bb1;
        Cmat[row1 + 32] = acc12[g] + bb2;
        Cmat[row1 + 48] = acc13[g] + bb3;
    }
}

// ---------------- K4: RoPE + head repack (q,k) ----------------
__global__ void rope_kernel(const float* __restrict__ proj, bf16_t* __restrict__ qb,
                            bf16_t* __restrict__ kbp) {
    size_t idx = (size_t)blockIdx.x * blockDim.x + threadIdx.x;   // BS_*H_*32 pairs
    if (idx >= (size_t)BS_ * 512) return;
    int p = (int)(idx & 31);
    int h = (int)((idx >> 5) & 15);
    size_t tokrow = idx >> 9;                   // b*S_ + s
    int s_tok = (int)(tokrow & (S_ - 1));
    int b = (int)(tokrow >> 12);
    int d = p * 2;
    float a = expf(-((float)p / 31.f) * 9.210340371976184f);  // 1/10000^linspace(0,1,32)
    float th = (float)s_tok * a;
    float sn = sinf(th), cn = cosf(th);
    const float* row = proj + tokrow * NPROJ_;
    float qe = row[h * 64 + d], qo = row[h * 64 + d + 1];
    float ke = row[1024 + h * 64 + d] * 0.125f, ko = row[1024 + h * 64 + d + 1] * 0.125f;
    size_t dst = ((size_t)(b * H_ + h) * S_ + s_tok) * DK_ + d;
    qb[dst]     = (bf16_t)(qe * cn - qo * sn);
    qb[dst + 1] = (bf16_t)(qo * cn + qe * sn);
    kbp[dst]     = (bf16_t)(ke * cn - ko * sn);
    kbp[dst + 1] = (bf16_t)(ko * cn + ke * sn);
}

// ---------------- K5: v repack ----------------
__global__ void vpack_kernel(const float* __restrict__ proj, bf16_t* __restrict__ vb) {
    size_t idx = (size_t)blockIdx.x * blockDim.x + threadIdx.x;   // BS_*2048
    if (idx >= (size_t)BS_ * 2048) return;
    int j = (int)(idx & 2047);
    size_t tokrow = idx >> 11;
    int h = j >> 7, d = j & 127;
    int b = (int)(tokrow >> 12);
    int s_tok = (int)(tokrow & (S_ - 1));
    vb[((size_t)(b * H_ + h) * S_ + s_tok) * DV_ + d] = (bf16_t)proj[tokrow * NPROJ_ + 2048 + j];
}

// ---------------- K6: per-chunk attention: inner, inner_scale, inner_out, KV ----------------
__global__ void attn_chunk_kernel(const bf16_t* __restrict__ qg, const bf16_t* __restrict__ kg,
                                  const bf16_t* __restrict__ vg, const float* __restrict__ mask_n,
                                  float* __restrict__ inner_out, float* __restrict__ inner_scale,
                                  float* __restrict__ KV) {
    const int c = blockIdx.x;
    const int bh = blockIdx.y;       // b*H + h
    const int h = bh & 15;
    __shared__ __attribute__((aligned(16))) bf16_t qs[64 * 64];
    __shared__ __attribute__((aligned(16))) bf16_t ks[64 * 64];
    __shared__ __attribute__((aligned(16))) bf16_t vs[64 * 128];
    __shared__ float ss[64 * 64];
    __shared__ float rsc[64];
    const int tid = threadIdx.x;
    const size_t qkoff = ((size_t)bh * S_ + (size_t)c * CL_) * DK_;
    const size_t voff  = ((size_t)bh * S_ + (size_t)c * CL_) * DV_;
    const uint4* q4 = (const uint4*)(qg + qkoff);
    const uint4* k4 = (const uint4*)(kg + qkoff);
    const uint4* v4 = (const uint4*)(vg + voff);
    uint4* qs4 = (uint4*)qs; uint4* ks4 = (uint4*)ks; uint4* vs4 = (uint4*)vs;
    for (int i = tid; i < 512; i += 256) { qs4[i] = q4[i]; ks4[i] = k4[i]; }
    for (int i = tid; i < 1024; i += 256) vs4[i] = v4[i];
    __syncthreads();

    const int lane = tid & 31, wave = tid >> 5;
    const int r = lane & 15, kb = (lane >> 4) << 3;
    const int dn = lane & 15, mh = (lane >> 4) << 3;

    // phase 1: inner = (q @ k^T) * mask_n   (16 tiles, 2 per wave)
    const float* mrow = mask_n + (size_t)h * 4096;
#pragma unroll
    for (int e = 0; e < 2; ++e) {
        int t = wave * 2 + e, ti = t >> 2, tj = t & 3;
        v8f acc = {};
        const bf16_t* arow = qs + (ti * 16 + r) * 64;
        const bf16_t* brow = ks + (tj * 16 + r) * 64;
#pragma unroll
        for (int k = 0; k < 64; k += 32)
            acc = wmma_bf16(load_frag_contig(arow + k, kb), load_frag_contig(brow + k, kb), acc);
#pragma unroll
        for (int g = 0; g < 8; ++g) {
            int i = ti * 16 + mh + g, j = tj * 16 + dn;
            ss[i * 64 + j] = acc[g] * mrow[i * 64 + j];
        }
    }
    __syncthreads();

    // phase 2: inner_scale[i] = max(|sum_j inner|, 1)
    if (tid < 64) {
        float s = 0.f;
#pragma unroll 8
        for (int j = 0; j < 64; ++j) s += ss[tid * 64 + j];
        float sc = fmaxf(fabsf(s), 1.f);
        inner_scale[((size_t)c * 32 + bh) * 64 + tid] = sc;
        rsc[tid] = 1.f / sc;
    }
    __syncthreads();

    // phases 3+4: inner_out = (inner/scale) @ v ; KV = k^T @ v   (32 tiles each, 4 per wave)
    const size_t iob = ((size_t)c * 32 + bh) * (DK_ * DV_);   // DK_*DV_ == CL_*DV_ == 8192
#pragma unroll
    for (int e = 0; e < 4; ++e) {
        int t = wave * 4 + e, ti = t >> 3, tn = t & 7;
        float sA = rsc[ti * 16 + r];
        v8f acc = {};
        v8f acc2 = {};
#pragma unroll
        for (int k = 0; k < 64; k += 32) {
            v16bf bv = load_frag_strided(vs + (size_t)(k + kb) * 128 + tn * 16 + r, 128);
            v16bf ai = load_frag_f32_scaled(&ss[(ti * 16 + r) * 64 + k + kb], sA);
            acc = wmma_bf16(ai, bv, acc);
            v16bf ak = load_frag_strided(ks + (size_t)(k + kb) * 64 + ti * 16 + r, 64);
            acc2 = wmma_bf16(ak, bv, acc2);
        }
#pragma unroll
        for (int g = 0; g < 8; ++g) {
            int i = ti * 16 + mh + g, d = tn * 16 + dn;
            inner_out[iob + (size_t)i * DV_ + d] = acc[g];
            KV[iob + (size_t)i * DV_ + d] = acc2[g];
        }
    }
}

// ---------------- K7: sequential chunk scan (in-place: KV -> cross_chunk) ----------------
__global__ void scan_kernel(float* __restrict__ KVc, float* __restrict__ cross_scale,
                            const float* __restrict__ cdec) {
    const int bh = blockIdx.x, h = bh & 15;
    const float cd = cdec[h];
    const int t = threadIdx.x;
    const int col = t & 127;
    float state[32];
#pragma unroll
    for (int rr = 0; rr < 32; ++rr) state[rr] = 0.f;
    __shared__ float part[256];
    __shared__ float scs[128];
    if (t < 128) scs[t] = 1.f;
    __syncthreads();
    for (int c = 0; c < NC_; ++c) {
        size_t base = ((size_t)c * 32 + bh) * 8192;
        float sc = scs[col];
        float inv = 1.f / sc;
        if (t < 128) cross_scale[((size_t)c * 32 + bh) * 128 + t] = sc;
        float myabs = 0.f;
#pragma unroll
        for (int rr = 0; rr < 32; ++rr) {
            size_t e = base + t + 256 * rr;
            float kv = KVc[e];
            KVc[e] = kv * inv;                    // cross = kv / kv_scale (written in place)
            state[rr] = state[rr] * cd + kv;      // new_chunk
            myabs += fabsf(state[rr]);
        }
        part[t] = myabs;
        __syncthreads();
        if (t < 128) scs[t] = fmaxf(part[t] + part[t + 128], 1.f);   // sum over dk of |new_chunk|
        __syncthreads();
    }
}

// ---------------- K8: cross_out + combine into token-major buffer ----------------
__global__ void cross_combine_kernel(const bf16_t* __restrict__ qg, const float* __restrict__ cross,
                                     const float* __restrict__ inner_out,
                                     const float* __restrict__ inner_scale,
                                     const float* __restrict__ cross_scale,
                                     const float* __restrict__ inner_decay,
                                     float* __restrict__ outc) {
    const int c = blockIdx.x, bh = blockIdx.y, h = bh & 15, b = bh >> 4;
    __shared__ __attribute__((aligned(16))) bf16_t cs[64 * 128];
    __shared__ float idec[64];
    __shared__ float risc[64];
    const int tid = threadIdx.x;
    const size_t cb = ((size_t)c * 32 + bh) * 8192;
    for (int i = tid; i < 8192; i += 256) cs[i] = (bf16_t)cross[cb + i];
    if (tid < 64) {
        idec[tid] = inner_decay[h * 64 + tid];
        risc[tid] = 1.f / inner_scale[((size_t)c * 32 + bh) * 64 + tid];
    }
    __syncthreads();
    const int lane = tid & 31, wave = tid >> 5;
    const int r = lane & 15, kb = (lane >> 4) << 3;
    const int dn = lane & 15, mh = (lane >> 4) << 3;
    const bf16_t* qc = qg + ((size_t)bh * S_ + (size_t)c * CL_) * DK_;
    const size_t csb = ((size_t)c * 32 + bh) * 128;
#pragma unroll
    for (int e = 0; e < 4; ++e) {
        int t = wave * 4 + e, ti = t >> 3, tn = t & 7;
        float sA = idec[ti * 16 + r];
        v8f acc = {};
#pragma unroll
        for (int k = 0; k < 64; k += 32) {
            v16bf a = load_frag_contig_scaled(qc + (ti * 16 + r) * 64 + k, kb, sA);
            v16bf bmat = load_frag_strided(cs + (size_t)(k + kb) * 128 + tn * 16 + r, 128);
            acc = wmma_bf16(a, bmat, acc);
        }
#pragma unroll
        for (int g = 0; g < 8; ++g) {
            int i = ti * 16 + mh + g, d = tn * 16 + dn;
            float val = inner_out[cb + (size_t)i * DV_ + d] / cross_scale[csb + d]
                      + acc[g] * risc[i];
            int s_tok = c * CL_ + i;
            outc[((size_t)b * S_ + s_tok) * NPROJ_ + h * DV_ + d] = val;
        }
    }
}

// ---------------- K9: GroupNorm + SiLU(gate) -> bf16 ----------------
__global__ void gn_gate_kernel(const float* __restrict__ proj, const float* __restrict__ gn_g,
                               const float* __restrict__ gn_b, bf16_t* __restrict__ y) {
    const int row = blockIdx.x, t = threadIdx.x;     // 8192 x 256 (8 ch/thread, 16 thr/group)
    const float* pr = proj + (size_t)row * NPROJ_;
    float vals[8]; float s = 0.f, q = 0.f;
#pragma unroll
    for (int i = 0; i < 8; ++i) { float v = pr[t * 8 + i]; vals[i] = v; s += v; q += v * v; }
    __shared__ float ps[256], pq[256];
    __shared__ float gmu[16], grs[16];
    ps[t] = s; pq[t] = q; __syncthreads();
    if ((t & 15) == 0) {
        float S = 0.f, Q = 0.f;
#pragma unroll
        for (int j = 0; j < 16; ++j) { S += ps[t + j]; Q += pq[t + j]; }
        float mu = S * (1.f / 128.f);
        float var = Q * (1.f / 128.f) - mu * mu;
        gmu[t >> 4] = mu; grs[t >> 4] = rsqrtf(var + 1e-5f);
    }
    __syncthreads();
    float mu = gmu[t >> 4], rsd = grs[t >> 4];
    bf16_t* yr = y + (size_t)row * 2048;
#pragma unroll
    for (int i = 0; i < 8; ++i) {
        int ch = t * 8 + i;
        float xg = (vals[i] - mu) * rsd * gn_g[ch] + gn_b[ch];
        float gt = pr[4096 + ch];
        float sig = gt / (1.f + expf(-gt));      // silu
        yr[ch] = (bf16_t)(sig * xg);
    }
}

// ---------------- host launcher ----------------
extern "C" void kernel_launch(void* const* d_in, const int* in_sizes, int n_in,
                              void* d_out, int out_size, void* d_ws, size_t ws_size,
                              hipStream_t stream) {
    const float* x      = (const float*)d_in[0];
    const float* amask  = (const float*)d_in[1];
    const float* W_q    = (const float*)d_in[2];
    const float* b_q    = (const float*)d_in[3];
    const float* W_k    = (const float*)d_in[4];
    const float* b_k    = (const float*)d_in[5];
    const float* W_v    = (const float*)d_in[6];
    const float* b_v    = (const float*)d_in[7];
    const float* W_g    = (const float*)d_in[8];
    const float* b_g    = (const float*)d_in[9];
    const float* W_o    = (const float*)d_in[10];
    const float* b_o    = (const float*)d_in[11];
    const float* ln_g   = (const float*)d_in[12];
    const float* ln_b   = (const float*)d_in[13];
    const float* gn_g   = (const float*)d_in[14];
    const float* gn_b   = (const float*)d_in[15];
    float* out = (float*)d_out;

    char* ws = (char*)d_ws;
    size_t off = 0;
    auto alloc = [&](size_t bytes) -> char* {
        off = (off + 255) & ~(size_t)255;
        char* p = ws + off;
        off += bytes;
        return p;
    };
    bf16_t* abf   = (bf16_t*)alloc((size_t)BS_ * 2048 * 2);      // xn (K=1024) then y (K=2048)
    bf16_t* wbf   = (bf16_t*)alloc((size_t)NPROJ_ * D_ * 2);     // packed weights (reused for W_o)
    float*  bias  = (float*)alloc((size_t)NPROJ_ * 4);
    float*  proj  = (float*)alloc((size_t)BS_ * NPROJ_ * 4);     // q|k|v|g, later out-combined
    bf16_t* qbf   = (bf16_t*)alloc((size_t)B_ * H_ * S_ * DK_ * 2);
    bf16_t* kbf   = (bf16_t*)alloc((size_t)B_ * H_ * S_ * DK_ * 2);
    bf16_t* vbf   = (bf16_t*)alloc((size_t)B_ * H_ * S_ * DV_ * 2);
    float*  iout  = (float*)alloc((size_t)NC_ * 32 * 8192 * 4);
    float*  iscl  = (float*)alloc((size_t)NC_ * 32 * 64 * 4);
    float*  kvbuf = (float*)alloc((size_t)NC_ * 32 * 8192 * 4);  // KV, then cross_chunk in place
    float*  cscl  = (float*)alloc((size_t)NC_ * 32 * 128 * 4);
    float*  maskn = (float*)alloc((size_t)H_ * 4096 * 4);
    float*  idec  = (float*)alloc((size_t)H_ * 64 * 4);
    float*  cdec  = (float*)alloc((size_t)H_ * 4);

    // constants
    prep_const_kernel<<<H_, 64, 0, stream>>>(amask, maskn, idec, cdec);

    // pack W_q|W_k|W_v|W_g (row-concatenated) + biases
    f32_to_bf16_kernel<<<4096, 256, 0, stream>>>(W_q, wbf, (size_t)D_ * D_);
    f32_to_bf16_kernel<<<4096, 256, 0, stream>>>(W_k, wbf + (size_t)1024 * D_, (size_t)D_ * D_);
    f32_to_bf16_kernel<<<8192, 256, 0, stream>>>(W_v, wbf + (size_t)2048 * D_, (size_t)2048 * D_);
    f32_to_bf16_kernel<<<8192, 256, 0, stream>>>(W_g, wbf + (size_t)4096 * D_, (size_t)2048 * D_);
    copy_f32_kernel<<<4, 256, 0, stream>>>(b_q, bias, 1024);
    copy_f32_kernel<<<4, 256, 0, stream>>>(b_k, bias + 1024, 1024);
    copy_f32_kernel<<<8, 256, 0, stream>>>(b_v, bias + 2048, 2048);
    copy_f32_kernel<<<8, 256, 0, stream>>>(b_g, bias + 4096, 2048);

    // layernorm -> bf16 activations
    layernorm_kernel<<<BS_, 256, 0, stream>>>(x, ln_g, ln_b, abf);

    // fused projection GEMM: (8192 x 1024) @ (6144 x 1024)^T
    gemm_bf16_kernel<<<dim3(NPROJ_ / 64, BS_ / 256), 256, 0, stream>>>(
        abf, wbf, bias, proj, BS_, NPROJ_, D_);

    // rope + repack
    rope_kernel<<<(unsigned)(((size_t)BS_ * 512 + 255) / 256), 256, 0, stream>>>(proj, qbf, kbf);
    vpack_kernel<<<(unsigned)(((size_t)BS_ * 2048 + 255) / 256), 256, 0, stream>>>(proj, vbf);

    // chunked attention
    attn_chunk_kernel<<<dim3(NC_, B_ * H_), 256, 0, stream>>>(
        qbf, kbf, vbf, maskn, iout, iscl, kvbuf);

    // sequential scan over chunks
    scan_kernel<<<B_ * H_, 256, 0, stream>>>(kvbuf, cscl, cdec);

    // cross term + combine (writes into proj cols [0,2048))
    cross_combine_kernel<<<dim3(NC_, B_ * H_), 256, 0, stream>>>(
        qbf, kvbuf, iout, iscl, cscl, idec, proj);

    // groupnorm + silu gate -> bf16 y (reuses abf)
    gn_gate_kernel<<<BS_, 256, 0, stream>>>(proj, gn_g, gn_b, abf);

    // output GEMM: (8192 x 2048) @ (1024 x 2048)^T -> d_out
    f32_to_bf16_kernel<<<8192, 256, 0, stream>>>(W_o, wbf, (size_t)D_ * 2048);
    copy_f32_kernel<<<4, 256, 0, stream>>>(b_o, bias, 1024);
    gemm_bf16_kernel<<<dim3(D_ / 64, BS_ / 256), 256, 0, stream>>>(
        abf, wbf, bias, out, BS_, D_, 2048);
}
